// CausalSelfAttention_32744830664994
// MI455X (gfx1250) — matrix operand order repaired and checked by reading, not compiled, not run on hardware
//
#include <hip/hip_runtime.h>

#define DEV __device__ __forceinline__

typedef __attribute__((ext_vector_type(16))) __bf16        v16bf;
typedef __attribute__((ext_vector_type(8)))  float         v8f;
typedef __attribute__((ext_vector_type(4)))  unsigned int  u32x4;

constexpr int Bsz = 4, T = 2048, C = 1024, H = 16, HD = 64, C3 = 3072;
constexpr int M = Bsz * T; // 8192 rows

struct Frag {
  union { v16bf v; u32x4 q[2]; unsigned int u[8]; };
};

DEV unsigned short bf16_rne(float f) {
  unsigned int u = __float_as_uint(f);
  u += 0x7FFFu + ((u >> 16) & 1u);
  return (unsigned short)(u >> 16);
}
DEV unsigned int pack2(float lo, float hi) {
  return (unsigned int)bf16_rne(lo) | ((unsigned int)bf16_rne(hi) << 16);
}
// Load a fragment as two 16-byte contiguous chunks at element offsets off0/off1.
DEV Frag load_frag2(const unsigned short* p, int off0, int off1) {
  Frag f;
  f.q[0] = *(const u32x4*)(p + off0);
  f.q[1] = *(const u32x4*)(p + off1);
  return f;
}
DEV v8f wmma_bf16(const Frag& a, const Frag& b, v8f c) {
  return __builtin_amdgcn_wmma_f32_16x16x32_bf16(false, a.v, false, b.v,
                                                 (short)0, c, false, false);
}

// ---------------- conversion kernels ----------------
__global__ void cvt_bf16(const float* __restrict__ in, unsigned short* __restrict__ out, int n) {
  int i = blockIdx.x * blockDim.x + threadIdx.x;
  if (i < n) out[i] = bf16_rne(in[i]);
}
// in: [K][N] row-major f32 -> out: [N][K] bf16 (transposed)
__global__ void cvt_bf16_t(const float* __restrict__ in, unsigned short* __restrict__ out,
                           int K, int N) {
  int i = blockIdx.x * blockDim.x + threadIdx.x;
  if (i < K * N) {
    int k = i / N, n = i - k * N;
    out[n * K + k] = bf16_rne(in[i]);
  }
}

// ---------------- QKV GEMM:  [M,C]bf16 x Wt[C3,C]bf16 + b -> q/k/vt buffers ----------------
__global__ __launch_bounds__(128) void gemm_qkv(
    const unsigned short* __restrict__ xb, const unsigned short* __restrict__ wt,
    const float* __restrict__ bias, unsigned short* __restrict__ qb,
    unsigned short* __restrict__ kbuf, unsigned short* __restrict__ vtb) {
  const int NT = C3 / 64; // 48 column tiles
  int wid  = (blockIdx.x * blockDim.x + threadIdx.x) >> 5;
  int lane = threadIdx.x & 31;
  int l15  = lane & 15;
  int hi   = lane >> 4;
  int h8   = hi << 3;   // C-frag M offset
  int h16  = hi << 4;   // B-frag kbase offset
  int m0 = (wid / NT) * 64;
  int n0 = (wid % NT) * 64;

  const unsigned short* arow[4];
  const unsigned short* brow[4];
#pragma unroll
  for (int i = 0; i < 4; ++i) {
    arow[i] = xb + (long)(m0 + i * 16 + l15) * C;
    brow[i] = wt + (long)(n0 + i * 16 + l15) * C;
  }
  v8f vzero = {0.f,0.f,0.f,0.f,0.f,0.f,0.f,0.f};
  v8f acc[4][4];
#pragma unroll
  for (int i = 0; i < 4; ++i)
#pragma unroll
    for (int j = 0; j < 4; ++j) acc[i][j] = vzero;

  for (int kk = 0; kk < C; kk += 32) {
    Frag a[4], b[4];
#pragma unroll
    for (int i = 0; i < 4; ++i) a[i] = load_frag2(arow[i], kk + h16, kk + h16 + 8);
#pragma unroll
    for (int i = 0; i < 4; ++i) b[i] = load_frag2(brow[i], kk + h16, kk + h16 + 8);
#pragma unroll
    for (int i = 0; i < 4; ++i)
#pragma unroll
      for (int j = 0; j < 4; ++j) acc[i][j] = wmma_bf16(a[i], b[j], acc[i][j]);
  }

#pragma unroll
  for (int j = 0; j < 4; ++j) {
    int ng = n0 + j * 16 + l15;
    float bv = bias[ng];
    int part = ng >> 10;       // 0=q 1=k 2=v
    int cc = ng & (C - 1);
    int h = cc >> 6;
    int d = cc & 63;
#pragma unroll
    for (int i = 0; i < 4; ++i) {
#pragma unroll
      for (int r = 0; r < 8; ++r) {
        int mg = m0 + i * 16 + h8 + r;
        int bi = mg >> 11;         // / T
        int t  = mg & (T - 1);
        float val = acc[i][j][r] + bv;
        if (part == 0) {
          qb[(((bi * H + h) * T + t) << 6) + d] = bf16_rne(val * 0.125f); // pre-scale 1/sqrt(hd)
        } else if (part == 1) {
          kbuf[(((bi * H + h) * T + t) << 6) + d] = bf16_rne(val);
        } else {
          vtb[(((bi * H + h) * HD + d) << 11) + t] = bf16_rne(val);       // V transposed
        }
      }
    }
  }
}

// ---------------- flash attention: one wave per (b,h,16-query tile) ----------------
__global__ __launch_bounds__(256) void attn(
    const unsigned short* __restrict__ qb, const unsigned short* __restrict__ kbuf,
    const unsigned short* __restrict__ vtb, unsigned short* __restrict__ yb) {
  int wid  = (blockIdx.x * blockDim.x + threadIdx.x) >> 5;
  int lane = threadIdx.x & 31;
  int l15  = lane & 15;
  int hi   = lane >> 4;
  int h8   = hi << 3;
  int h16  = hi << 4;
  int qt = wid & (T / 16 - 1);  // 128 query tiles
  int bh = wid >> 7;            // b*H + h
  int q0 = qt << 4;

  const unsigned short* Qrow = qb  + (long)(bh * T + q0 + l15) * HD; // row for B-frag column
  const unsigned short* Kb   = kbuf + (long)bh * T * HD;
  const unsigned short* Vt   = vtb  + (long)bh * HD * T;

  // Q^T B-fragments (persistent): hd 0..31 and 32..63
  Frag bq0 = load_frag2(Qrow, h8, h8 + 16);
  Frag bq1 = load_frag2(Qrow, 32 + h8, 48 + h8);

  v8f vzero = {0.f,0.f,0.f,0.f,0.f,0.f,0.f,0.f};
  v8f ot[4];
#pragma unroll
  for (int t = 0; t < 4; ++t) ot[t] = vzero;
  float m_run = -1e30f, l_run = 0.f;
  int qg = q0 + l15;            // this lane's query index

  for (int j0 = 0; j0 < q0 + 16; j0 += 32) {
    // S^T = K x Q^T : X covers keys j0..j0+15, Y covers j0+16..j0+31
    const unsigned short* Kr0 = Kb + (long)(j0 + l15) * HD;
    const unsigned short* Kr1 = Kb + (long)(j0 + 16 + l15) * HD;
    v8f X = vzero, Y = vzero;
    Frag ak0 = load_frag2(Kr0, h8, h8 + 16);
    Frag ak1 = load_frag2(Kr0, 32 + h8, 48 + h8);
    X = wmma_bf16(ak0, bq0, X);
    X = wmma_bf16(ak1, bq1, X);
    Frag ak2 = load_frag2(Kr1, h8, h8 + 16);
    Frag ak3 = load_frag2(Kr1, 32 + h8, 48 + h8);
    Y = wmma_bf16(ak2, bq0, Y);
    Y = wmma_bf16(ak3, bq1, Y);

    // causal mask (lane = query, regs = keys)
    float xs[8], ys[8];
#pragma unroll
    for (int r = 0; r < 8; ++r) {
      int ktx = j0 + h8 + r;
      int kty = j0 + 16 + h8 + r;
      xs[r] = (ktx <= qg) ? X[r] : -1e30f;
      ys[r] = (kty <= qg) ? Y[r] : -1e30f;
    }
    // online softmax: per-lane reg reduction + merge key-halves across lane 16 swap
    float bm = xs[0];
#pragma unroll
    for (int r = 1; r < 8; ++r) bm = fmaxf(bm, xs[r]);
#pragma unroll
    for (int r = 0; r < 8; ++r) bm = fmaxf(bm, ys[r]);
    bm = fmaxf(bm, __shfl_xor(bm, 16, 32));
    float m_new = fmaxf(m_run, bm);
    float corr = __expf(m_run - m_new);
    float psum = 0.f;
    float px[8], py[8];
#pragma unroll
    for (int r = 0; r < 8; ++r) { px[r] = __expf(xs[r] - m_new); psum += px[r]; }
#pragma unroll
    for (int r = 0; r < 8; ++r) { py[r] = __expf(ys[r] - m_new); psum += py[r]; }
    psum += __shfl_xor(psum, 16, 32);
    l_run = l_run * corr + psum;
    m_run = m_new;

    // Build P^T B-fragment: B lanes 0-15 need keys 0..15, lanes 16-31 keys 16..31.
    unsigned int xk[4], yk[4];
#pragma unroll
    for (int i = 0; i < 4; ++i) {
      xk[i] = pack2(px[2 * i], px[2 * i + 1]);
      yk[i] = pack2(py[2 * i], py[2 * i + 1]);
    }
    Frag bp;
#pragma unroll
    for (int i = 0; i < 4; ++i) {
      unsigned int sx = (unsigned int)__shfl_xor((int)xk[i], 16, 32);
      unsigned int sy = (unsigned int)__shfl_xor((int)yk[i], 16, 32);
      bp.u[i]     = hi ? sy : xk[i];  // keys 0..7   | 16..23
      bp.u[4 + i] = hi ? yk[i] : sx;  // keys 8..15  | 24..31
    }
    // O^T += V^T x P^T (per-lane rescale: lane = query)
#pragma unroll
    for (int t = 0; t < 4; ++t) {
      const unsigned short* Vr = Vt + (long)(t * 16 + l15) * T;
      Frag av = load_frag2(Vr, j0 + h16, j0 + h16 + 8);
#pragma unroll
      for (int r = 0; r < 8; ++r) ot[t][r] *= corr;
      ot[t] = wmma_bf16(av, bp, ot[t]);
    }
  }

  float inv_l = __fdividef(1.f, l_run);
  int bi = bh >> 4, h = bh & 15;
  unsigned short* yrow = yb + ((long)(bi * T + qg) * C + h * HD);
#pragma unroll
  for (int t = 0; t < 4; ++t) {
    u32x4 o;
    o.x = pack2(ot[t][0] * inv_l, ot[t][1] * inv_l);
    o.y = pack2(ot[t][2] * inv_l, ot[t][3] * inv_l);
    o.z = pack2(ot[t][4] * inv_l, ot[t][5] * inv_l);
    o.w = pack2(ot[t][6] * inv_l, ot[t][7] * inv_l);
    *(u32x4*)(yrow + t * 16 + h8) = o;   // 8 consecutive hd elements, 16B store
  }
}

// ---------------- proj GEMM: y[M,C]bf16 x Wt[C,C]bf16 + b -> out f32 ----------------
__global__ __launch_bounds__(128) void gemm_proj(
    const unsigned short* __restrict__ yb, const unsigned short* __restrict__ wt,
    const float* __restrict__ bias, float* __restrict__ out) {
  const int NT = C / 64; // 16
  int wid  = (blockIdx.x * blockDim.x + threadIdx.x) >> 5;
  int lane = threadIdx.x & 31;
  int l15  = lane & 15;
  int hi   = lane >> 4;
  int h8   = hi << 3;
  int h16  = hi << 4;
  int m0 = (wid / NT) * 64;
  int n0 = (wid % NT) * 64;

  const unsigned short* arow[4];
  const unsigned short* brow[4];
#pragma unroll
  for (int i = 0; i < 4; ++i) {
    arow[i] = yb + (long)(m0 + i * 16 + l15) * C;
    brow[i] = wt + (long)(n0 + i * 16 + l15) * C;
  }
  v8f vzero = {0.f,0.f,0.f,0.f,0.f,0.f,0.f,0.f};
  v8f acc[4][4];
#pragma unroll
  for (int i = 0; i < 4; ++i)
#pragma unroll
    for (int j = 0; j < 4; ++j) acc[i][j] = vzero;

  for (int kk = 0; kk < C; kk += 32) {
    Frag a[4], b[4];
#pragma unroll
    for (int i = 0; i < 4; ++i) a[i] = load_frag2(arow[i], kk + h16, kk + h16 + 8);
#pragma unroll
    for (int i = 0; i < 4; ++i) b[i] = load_frag2(brow[i], kk + h16, kk + h16 + 8);
#pragma unroll
    for (int i = 0; i < 4; ++i)
#pragma unroll
      for (int j = 0; j < 4; ++j) acc[i][j] = wmma_bf16(a[i], b[j], acc[i][j]);
  }

#pragma unroll
  for (int j = 0; j < 4; ++j) {
    int ng = n0 + j * 16 + l15;
    float bv = bias[ng];
#pragma unroll
    for (int i = 0; i < 4; ++i) {
#pragma unroll
      for (int r = 0; r < 8; ++r) {
        int mg = m0 + i * 16 + h8 + r;
        out[(long)mg * C + ng] = acc[i][j][r] + bv;
      }
    }
  }
}

// ---------------- launch ----------------
extern "C" void kernel_launch(void* const* d_in, const int* in_sizes, int n_in,
                              void* d_out, int out_size, void* d_ws, size_t ws_size,
                              hipStream_t stream) {
  const float* x      = (const float*)d_in[0];
  const float* W_attn = (const float*)d_in[1];
  const float* b_attn = (const float*)d_in[2];
  const float* W_proj = (const float*)d_in[3];
  const float* b_proj = (const float*)d_in[4];
  float* out = (float*)d_out;

  char* ws = (char*)d_ws;
  size_t off = 0;
  auto alloc = [&](size_t bytes) -> char* {
    char* p = ws + off;
    off += (bytes + 255) & ~(size_t)255;
    return p;
  };
  unsigned short* xb  = (unsigned short*)alloc((size_t)M * C * 2);
  unsigned short* wta = (unsigned short*)alloc((size_t)C3 * C * 2);
  unsigned short* wtp = (unsigned short*)alloc((size_t)C * C * 2);
  unsigned short* qbf = (unsigned short*)alloc((size_t)Bsz * H * T * HD * 2);
  unsigned short* kbf = (unsigned short*)alloc((size_t)Bsz * H * T * HD * 2);
  unsigned short* vtb = (unsigned short*)alloc((size_t)Bsz * H * HD * T * 2);
  unsigned short* yb  = (unsigned short*)alloc((size_t)M * C * 2);

  int nx = M * C;
  cvt_bf16<<<(nx + 255) / 256, 256, 0, stream>>>(x, xb, nx);
  cvt_bf16_t<<<(C * C3 + 255) / 256, 256, 0, stream>>>(W_attn, wta, C, C3);
  cvt_bf16_t<<<(C * C + 255) / 256, 256, 0, stream>>>(W_proj, wtp, C, C);

  gemm_qkv<<<(M / 64) * (C3 / 64) / 4, 128, 0, stream>>>(xb, wta, b_attn, qbf, kbf, vtb);
  attn<<<(Bsz * H * (T / 16)) / 8, 256, 0, stream>>>(qbf, kbf, vtb, yb);
  gemm_proj<<<(M / 64) * (C / 64) / 4, 128, 0, stream>>>(yb, wtp, b_proj, out);
}
